// GATModel_68186900792262
// MI455X (gfx1250) — matrix operand order repaired and checked by reading, not compiled, not run on hardware
//
#include <hip/hip_runtime.h>

typedef __attribute__((ext_vector_type(16))) _Float16 v16h;
typedef __attribute__((ext_vector_type(8)))  _Float16 v8h;
typedef __attribute__((ext_vector_type(4)))  _Float16 v4h;
typedef __attribute__((ext_vector_type(8)))  float    v8f;

#define NN    50000
#define NE    800000
#define FIN   512
#define HIDC  64
#define NHEAD 4
#define NCLS  40
#define NSLOPE 0.2f
#define ET    (NE + NN)
#define HF    (NHEAD * HIDC)   /* 256 */

// ---------------- helpers ----------------

__device__ __forceinline__ float lrelu(float v) { return v < 0.f ? NSLOPE * v : v; }

// monotonic float -> uint key (order-preserving), for atomicMax-based segment max
__device__ __forceinline__ unsigned fkey(float x) {
    unsigned b = __float_as_uint(x);
    return (b & 0x80000000u) ? ~b : (b | 0x80000000u);
}
__device__ __forceinline__ float funkey(unsigned k) {
    unsigned b = (k & 0x80000000u) ? (k & 0x7FFFFFFFu) : ~k;
    return __uint_as_float(b);
}

__device__ __forceinline__ v8f v8f_zero() {
    v8f z;
#pragma unroll
    for (int t = 0; t < 8; ++t) z[t] = 0.f;
    return z;
}

// async DMA: global (16B per lane) -> LDS, tracked by ASYNCcnt
__device__ __forceinline__ void async_b128(unsigned lds_addr, unsigned long long gaddr) {
    asm volatile("global_load_async_to_lds_b128 %0, %1, off"
                 :: "v"(lds_addr), "v"(gaddr) : "memory");
}
__device__ __forceinline__ void wait_async0() {
    asm volatile("s_wait_asynccnt 0x0" ::: "memory");
}
__device__ __forceinline__ unsigned lds_off(const _Float16* p) {
    return (unsigned)(size_t)p;   // LDS aperture: hw uses addr[31:0]
}

// A fragment 16x32 f16 (ISA 7.12.2): lane m=l&15, g=l>>4; VGPR v slot j -> k=(v/4)*16+g*8+(v%4)*2+j
__device__ __forceinline__ v16h frag_a(const _Float16* tile, int row, int g) {
    const _Float16* p = tile + row * 40;
    v8h lo = *reinterpret_cast<const v8h*>(p + g * 8);
    v8h hi = *reinterpret_cast<const v8h*>(p + 16 + g * 8);
    v16h r;
#pragma unroll
    for (int t = 0; t < 8; ++t) { r[t] = lo[t]; r[8 + t] = hi[t]; }
    return r;
}

// B fragment 32x16 f16: lane n=l&15, g=l>>4; halves 0..7 -> k = g*8 + 0..7, 8..15 -> k = 16 + g*8 + 0..7 (tile stored transposed [n][k])
__device__ __forceinline__ v16h frag_b(const _Float16* tile, int col, int g) {
    const _Float16* p = tile + col * 40;
    v8h lo = *reinterpret_cast<const v8h*>(p + g * 8);
    v8h hi = *reinterpret_cast<const v8h*>(p + 16 + g * 8);
    v16h r;
#pragma unroll
    for (int t = 0; t < 8; ++t) { r[t] = lo[t]; r[8 + t] = hi[t]; }
    return r;
}

__device__ __forceinline__ void edge_sd(const int* __restrict__ ei, int e, int& s, int& d) {
    if (e < NE) { s = ei[e]; d = ei[NE + e]; }
    else        { s = e - NE; d = e - NE; }
}

// ---------------- x -> f16 convert ----------------
__global__ void k_xconv(const float* __restrict__ x, _Float16* __restrict__ xh) {
    int i = blockIdx.x * blockDim.x + threadIdx.x;   // i indexes groups of 4
    if (i >= (NN * FIN) / 4) return;
    float4 f = reinterpret_cast<const float4*>(x)[i];
    v4h o;
    o[0] = (_Float16)f.x; o[1] = (_Float16)f.y;
    o[2] = (_Float16)f.z; o[3] = (_Float16)f.w;
    reinterpret_cast<v4h*>(xh)[i] = o;
}

// ---------------- weight prep: transpose + f16 convert ----------------
// w1t: [256][512] f16 = W1^T ; w2t: [48][256] f16 = W2^T zero-padded to 48 cols
__global__ void k_convert(const float* __restrict__ W1, const float* __restrict__ W2,
                          _Float16* __restrict__ w1t, _Float16* __restrict__ w2t) {
    int i = blockIdx.x * blockDim.x + threadIdx.x;
    const int tot1 = HF * FIN;           // 131072
    if (i < tot1) {
        int c = i / FIN, k = i % FIN;
        w1t[i] = (_Float16)W1[(size_t)k * HF + c];
    }
    int i2 = i - tot1;
    if (i2 >= 0 && i2 < 48 * HF) {
        int n = i2 / HF, k = i2 % HF;
        w2t[i2] = (n < NCLS) ? (_Float16)W2[(size_t)k * NCLS + n] : (_Float16)0.0f;
    }
}

// ---------------- init scratch (harness poisons ws/out) ----------------
__global__ void k_init(float* agg1, unsigned* m1k, float* den1,
                       unsigned* m2k, float* den2,
                       float* out, const float* __restrict__ b2) {
    int stride = gridDim.x * blockDim.x;
    int i0 = blockIdx.x * blockDim.x + threadIdx.x;
    for (int i = i0; i < NN * HF; i += stride) agg1[i] = 0.f;
    for (int i = i0; i < NN * NHEAD; i += stride) { m1k[i] = 0u; den1[i] = 0.f; }
    for (int i = i0; i < NN; i += stride) { m2k[i] = 0u; den2[i] = 0.f; }
    for (int i = i0; i < NN * NCLS; i += stride) out[i] = b2[i % NCLS];
}

// ---------------- GEMM1: h1h[N,256] = f16( xh[N,512] @ W1 ) ----------------
// BM=128, BN=256 (full width), double-buffered async global->LDS pipeline.
__global__ __launch_bounds__(256) void k_gemm1(const _Float16* __restrict__ xh,
                                               const _Float16* __restrict__ w1t,
                                               _Float16* __restrict__ h1h) {
    __shared__ __align__(16) _Float16 sA[2][128 * 40];
    __shared__ __align__(16) _Float16 sB[2][256 * 40];
    const int tid = threadIdx.x;
    const int gm = blockIdx.x * 128;
    const int lane = tid & 31, wave = tid >> 5;
    const int wr = wave >> 1;            // 0..3 : 32-row slice
    const int wc = wave & 1;             // 0..1 : 128-col slice
    const int g = lane >> 4, ln = lane & 15;

    // per-thread staging assignment
    const int r = tid >> 1;              // A tile row 0..127
    const int halfsel = (tid & 1) * 16;  // halves 0..15 / 16..31
    const int rowg = gm + r;
    const int rowc = rowg < NN ? rowg : (NN - 1);   // clamp; junk rows never written back
    const _Float16* aSrc = xh + (size_t)rowc * FIN + halfsel;
    const _Float16* bSrc = w1t + (size_t)tid * FIN; // n = tid (0..255)

    v8f acc[2][8];
#pragma unroll
    for (int i = 0; i < 2; ++i)
#pragma unroll
        for (int j = 0; j < 8; ++j) acc[i][j] = v8f_zero();

    auto issue = [&](int kk, int buf) {
        unsigned la = lds_off(&sA[buf][r * 40 + halfsel]);
        unsigned long long ga = (unsigned long long)(aSrc + kk);
        async_b128(la, ga);
        async_b128(la + 16, ga + 16);
        unsigned lb = lds_off(&sB[buf][tid * 40]);
        unsigned long long gb = (unsigned long long)(bSrc + kk);
        async_b128(lb,      gb);
        async_b128(lb + 16, gb + 16);
        async_b128(lb + 32, gb + 32);
        async_b128(lb + 48, gb + 48);
    };

    issue(0, 0);
    const int NK = FIN / 32;   // 16
    for (int it = 0; it < NK; ++it) {
        const int buf = it & 1;
        wait_async0();
        __syncthreads();                       // everyone's tile landed; prev buffer free
        if (it + 1 < NK) issue((it + 1) * 32, buf ^ 1);

        v16h afr[2];
        afr[0] = frag_a(sA[buf], wr * 32 + ln, g);
        afr[1] = frag_a(sA[buf], wr * 32 + 16 + ln, g);
#pragma unroll
        for (int j = 0; j < 8; ++j) {
            v16h b = frag_b(sB[buf], wc * 128 + j * 16 + ln, g);
            acc[0][j] = __builtin_amdgcn_wmma_f32_16x16x32_f16(
                false, afr[0], false, b, (short)0, acc[0][j], false, false);
            acc[1][j] = __builtin_amdgcn_wmma_f32_16x16x32_f16(
                false, afr[1], false, b, (short)0, acc[1][j], false, false);
        }
    }

#pragma unroll
    for (int i = 0; i < 2; ++i)
#pragma unroll
        for (int j = 0; j < 8; ++j) {
            int col = wc * 128 + j * 16 + ln;
#pragma unroll
            for (int v = 0; v < 8; ++v) {
                int row = gm + wr * 32 + i * 16 + g * 8 + v;
                if (row < NN) h1h[(size_t)row * HF + col] = (_Float16)acc[i][j][v];
            }
        }
}

// ---------------- attention dots layer 1 ----------------
__global__ void k_dots1(const _Float16* __restrict__ h1h,
                        const float* __restrict__ aw_s, const float* __restrict__ aw_d,
                        float* __restrict__ as1, float* __restrict__ ad1) {
    int n = blockIdx.x * blockDim.x + threadIdx.x;
    if (n >= NN) return;
    const _Float16* hp = h1h + (size_t)n * HF;
#pragma unroll
    for (int h = 0; h < NHEAD; ++h) {
        float s = 0.f, d = 0.f;
        for (int c = 0; c < HIDC; ++c) {
            float v = (float)hp[h * HIDC + c];
            s += v * aw_s[h * HIDC + c];
            d += v * aw_d[h * HIDC + c];
        }
        as1[n * NHEAD + h] = s;
        ad1[n * NHEAD + h] = d;
    }
}

// ---------------- edge passes layer 1 ----------------
__global__ void k_emax1(const int* __restrict__ ei, const float* __restrict__ as1,
                        const float* __restrict__ ad1, unsigned* __restrict__ m1k) {
    int e = blockIdx.x * blockDim.x + threadIdx.x;
    if (e >= ET) return;
    int s, d; edge_sd(ei, e, s, d);
#pragma unroll
    for (int h = 0; h < NHEAD; ++h) {
        float v = lrelu(as1[s * NHEAD + h] + ad1[d * NHEAD + h]);
        atomicMax(&m1k[d * NHEAD + h], fkey(v));
    }
}

__global__ void k_esum1(const int* __restrict__ ei, const float* __restrict__ as1,
                        const float* __restrict__ ad1, const unsigned* __restrict__ m1k,
                        float* __restrict__ den1) {
    int e = blockIdx.x * blockDim.x + threadIdx.x;
    if (e >= ET) return;
    int s, d; edge_sd(ei, e, s, d);
#pragma unroll
    for (int h = 0; h < NHEAD; ++h) {
        float v = lrelu(as1[s * NHEAD + h] + ad1[d * NHEAD + h]);
        float ex = __expf(v - funkey(m1k[d * NHEAD + h]));
        atomicAdd(&den1[d * NHEAD + h], ex);
    }
}

// one wave per edge; lane covers 8 of the 256 feature slots
__global__ __launch_bounds__(256) void k_eagg1(const int* __restrict__ ei,
                                               const float* __restrict__ ew,
                                               const float* __restrict__ as1,
                                               const float* __restrict__ ad1,
                                               const unsigned* __restrict__ m1k,
                                               const float* __restrict__ den1,
                                               const _Float16* __restrict__ h1h,
                                               float* __restrict__ agg1) {
    int e = blockIdx.x * 8 + (threadIdx.x >> 5);
    int lane = threadIdx.x & 31;
    if (e >= ET) return;
    int s, d; edge_sd(ei, e, s, d);
    float w = (e < NE) ? ew[e] : 1.0f;
    float alpha[NHEAD];
#pragma unroll
    for (int h = 0; h < NHEAD; ++h) {
        float v = lrelu(as1[s * NHEAD + h] + ad1[d * NHEAD + h]);
        float ex = __expf(v - funkey(m1k[d * NHEAD + h]));
        alpha[h] = ex / (den1[d * NHEAD + h] + 1e-16f) * w;
    }
    const v8h hv = *reinterpret_cast<const v8h*>(h1h + (size_t)s * HF + lane * 8);
    float* dstp = agg1 + (size_t)d * HF;
#pragma unroll
    for (int u = 0; u < 8; ++u) {
        int idx = lane * 8 + u;
        atomicAdd(&dstp[idx], (float)hv[u] * alpha[idx >> 6]);
    }
}

// ---------------- relu + f16 convert for layer-2 input ----------------
__global__ void k_relu(const float* __restrict__ agg1, const float* __restrict__ b1,
                       _Float16* __restrict__ h2in) {
    int i = blockIdx.x * blockDim.x + threadIdx.x;
    if (i >= NN * HF) return;
    float v = agg1[i] + b1[i & (HF - 1)];
    h2in[i] = (_Float16)(v > 0.f ? v : 0.f);
}

// ---------------- GEMM2: h2h[N,40] = f16( h2in[N,256] @ W2 ) ----------------
// BM=128, 3 n-tiles (48 padded cols), double-buffered async pipeline.
__global__ __launch_bounds__(256) void k_gemm2(const _Float16* __restrict__ h2in,
                                               const _Float16* __restrict__ w2t,
                                               _Float16* __restrict__ h2h) {
    __shared__ __align__(16) _Float16 sA[2][128 * 40];
    __shared__ __align__(16) _Float16 sB[2][48 * 40];
    const int tid = threadIdx.x;
    const int gm = blockIdx.x * 128;
    const int lane = tid & 31, wave = tid >> 5;
    const int g = lane >> 4, ln = lane & 15;

    const int r = tid >> 1;
    const int halfsel = (tid & 1) * 16;
    const int rowg = gm + r;
    const int rowc = rowg < NN ? rowg : (NN - 1);
    const _Float16* aSrc = h2in + (size_t)rowc * HF + halfsel;
    const bool bldr = tid < 96;                       // 48 rows x 2 halves
    const _Float16* bSrc = w2t + (size_t)(tid >> 1) * HF + halfsel;

    v8f acc[3];
#pragma unroll
    for (int j = 0; j < 3; ++j) acc[j] = v8f_zero();

    auto issue = [&](int kk, int buf) {
        unsigned la = lds_off(&sA[buf][r * 40 + halfsel]);
        unsigned long long ga = (unsigned long long)(aSrc + kk);
        async_b128(la, ga);
        async_b128(la + 16, ga + 16);
        if (bldr) {
            unsigned lb = lds_off(&sB[buf][(tid >> 1) * 40 + halfsel]);
            unsigned long long gb = (unsigned long long)(bSrc + kk);
            async_b128(lb, gb);
            async_b128(lb + 16, gb + 16);
        }
    };

    issue(0, 0);
    const int NK = HF / 32;   // 8
    for (int it = 0; it < NK; ++it) {
        const int buf = it & 1;
        wait_async0();
        __syncthreads();
        if (it + 1 < NK) issue((it + 1) * 32, buf ^ 1);

        v16h a = frag_a(sA[buf], wave * 16 + ln, g);
#pragma unroll
        for (int j = 0; j < 3; ++j) {
            v16h b = frag_b(sB[buf], j * 16 + ln, g);
            acc[j] = __builtin_amdgcn_wmma_f32_16x16x32_f16(
                false, a, false, b, (short)0, acc[j], false, false);
        }
    }

#pragma unroll
    for (int j = 0; j < 3; ++j) {
        int col = j * 16 + ln;
        if (col >= NCLS) continue;
#pragma unroll
        for (int v = 0; v < 8; ++v) {
            int row = gm + wave * 16 + g * 8 + v;
            if (row < NN) h2h[(size_t)row * NCLS + col] = (_Float16)acc[j][v];
        }
    }
}

// ---------------- attention dots layer 2 (heads = 1) ----------------
__global__ void k_dots2(const _Float16* __restrict__ h2h,
                        const float* __restrict__ aw_s, const float* __restrict__ aw_d,
                        float* __restrict__ as2, float* __restrict__ ad2) {
    int n = blockIdx.x * blockDim.x + threadIdx.x;
    if (n >= NN) return;
    const _Float16* hp = h2h + (size_t)n * NCLS;
    float s = 0.f, d = 0.f;
    for (int c = 0; c < NCLS; ++c) {
        float v = (float)hp[c];
        s += v * aw_s[c];
        d += v * aw_d[c];
    }
    as2[n] = s; ad2[n] = d;
}

// ---------------- edge passes layer 2 ----------------
__global__ void k_emax2(const int* __restrict__ ei, const float* __restrict__ as2,
                        const float* __restrict__ ad2, unsigned* __restrict__ m2k) {
    int e = blockIdx.x * blockDim.x + threadIdx.x;
    if (e >= ET) return;
    int s, d; edge_sd(ei, e, s, d);
    float v = lrelu(as2[s] + ad2[d]);
    atomicMax(&m2k[d], fkey(v));
}

__global__ void k_esum2(const int* __restrict__ ei, const float* __restrict__ as2,
                        const float* __restrict__ ad2, const unsigned* __restrict__ m2k,
                        float* __restrict__ den2) {
    int e = blockIdx.x * blockDim.x + threadIdx.x;
    if (e >= ET) return;
    int s, d; edge_sd(ei, e, s, d);
    float v = lrelu(as2[s] + ad2[d]);
    atomicAdd(&den2[d], __expf(v - funkey(m2k[d])));
}

__global__ __launch_bounds__(256) void k_eagg2(const int* __restrict__ ei,
                                               const float* __restrict__ ew,
                                               const float* __restrict__ as2,
                                               const float* __restrict__ ad2,
                                               const unsigned* __restrict__ m2k,
                                               const float* __restrict__ den2,
                                               const _Float16* __restrict__ h2h,
                                               float* __restrict__ out) {
    int e = blockIdx.x * 8 + (threadIdx.x >> 5);
    int lane = threadIdx.x & 31;
    if (e >= ET) return;
    int s, d; edge_sd(ei, e, s, d);
    float w = (e < NE) ? ew[e] : 1.0f;
    float v = lrelu(as2[s] + ad2[d]);
    float ex = __expf(v - funkey(m2k[d]));
    float alpha = ex / (den2[d] + 1e-16f) * w;
    const _Float16* hp = h2h + (size_t)s * NCLS;
    float* op = out + (size_t)d * NCLS;
    if (lane < NCLS) atomicAdd(&op[lane], (float)hp[lane] * alpha);
    int c2 = lane + 32;
    if (c2 < NCLS)   atomicAdd(&op[c2],   (float)hp[c2]   * alpha);
}

// ---------------- host launcher ----------------
extern "C" void kernel_launch(void* const* d_in, const int* in_sizes, int n_in,
                              void* d_out, int out_size, void* d_ws, size_t ws_size,
                              hipStream_t stream) {
    const float* x    = (const float*)d_in[0];
    const int*   ei   = (const int*)d_in[1];
    const float* ew   = (const float*)d_in[2];
    const float* W1   = (const float*)d_in[3];
    const float* a_s1 = (const float*)d_in[4];
    const float* a_d1 = (const float*)d_in[5];
    const float* b1   = (const float*)d_in[6];
    const float* W2   = (const float*)d_in[7];
    const float* a_s2 = (const float*)d_in[8];
    const float* a_d2 = (const float*)d_in[9];
    const float* b2   = (const float*)d_in[10];
    float* out = (float*)d_out;

    char* p = (char*)d_ws;
    auto alloc = [&](size_t bytes) -> char* {
        char* r = p;
        p += (bytes + 255) & ~(size_t)255;
        return r;
    };
    _Float16* xh   = (_Float16*)alloc((size_t)NN * FIN * 2);
    _Float16* w1t  = (_Float16*)alloc((size_t)HF * FIN * 2);
    _Float16* w2t  = (_Float16*)alloc((size_t)48 * HF * 2);
    _Float16* h1h  = (_Float16*)alloc((size_t)NN * HF * 2);
    float*    as1  = (float*)   alloc((size_t)NN * NHEAD * 4);
    float*    ad1  = (float*)   alloc((size_t)NN * NHEAD * 4);
    unsigned* m1k  = (unsigned*)alloc((size_t)NN * NHEAD * 4);
    float*    den1 = (float*)   alloc((size_t)NN * NHEAD * 4);
    float*    agg1 = (float*)   alloc((size_t)NN * HF * 4);
    _Float16* h2in = (_Float16*)alloc((size_t)NN * HF * 2);
    _Float16* h2h  = (_Float16*)alloc((size_t)NN * NCLS * 2);
    float*    as2  = (float*)   alloc((size_t)NN * 4);
    float*    ad2  = (float*)   alloc((size_t)NN * 4);
    unsigned* m2k  = (unsigned*)alloc((size_t)NN * 4);
    float*    den2 = (float*)   alloc((size_t)NN * 4);

    k_xconv<<<(NN * FIN / 4 + 255) / 256, 256, 0, stream>>>(x, xh);

    const int cvt_tot = HF * FIN + 48 * HF;
    k_convert<<<(cvt_tot + 255) / 256, 256, 0, stream>>>(W1, W2, w1t, w2t);
    k_init<<<4096, 256, 0, stream>>>(agg1, m1k, den1, m2k, den2, out, b2);

    k_gemm1<<<(NN + 127) / 128, 256, 0, stream>>>(xh, w1t, h1h);

    k_dots1<<<(NN + 255) / 256, 256, 0, stream>>>(h1h, a_s1, a_d1, as1, ad1);

    const int eb = (ET + 255) / 256;
    const int wb = (ET + 7) / 8;
    k_emax1<<<eb, 256, 0, stream>>>(ei, as1, ad1, m1k);
    k_esum1<<<eb, 256, 0, stream>>>(ei, as1, ad1, m1k, den1);
    k_eagg1<<<wb, 256, 0, stream>>>(ei, ew, as1, ad1, m1k, den1, h1h, agg1);

    k_relu<<<(NN * HF + 255) / 256, 256, 0, stream>>>(agg1, b1, h2in);

    k_gemm2<<<(NN + 127) / 128, 256, 0, stream>>>(h2in, w2t, h2h);

    k_dots2<<<(NN + 255) / 256, 256, 0, stream>>>(h2h, a_s2, a_d2, as2, ad2);
    k_emax2<<<eb, 256, 0, stream>>>(ei, as2, ad2, m2k);
    k_esum2<<<eb, 256, 0, stream>>>(ei, as2, ad2, m2k, den2);
    k_eagg2<<<wb, 256, 0, stream>>>(ei, ew, as2, ad2, m2k, den2, h2h, out);
}
